// MultiHeadFusionGRF_65197603553329
// MI455X (gfx1250) — matrix operand order repaired and checked by reading, not compiled, not run on hardware
//
#include <hip/hip_runtime.h>
#include <hip/hip_bf16.h>
#include <math.h>

// ---------------------------------------------------------------------------
// MultiHeadFusionGRF for MI455X (gfx1250, wave32, WMMA 16x16x32 f16 -> f32)
// B=65536, S=4, D=E=128, H=4, HD=32
// 2 waves/block, each wave owns 16 batch rows; ~81.5KB LDS per wave so
// 2 blocks (4 waves) co-reside per WGP (320KB LDS).
// LDS row pitches padded (+2 elements) so row-parallel accesses hit
// distinct banks (stride 128 elems would alias all rows to one bank).
// ---------------------------------------------------------------------------

typedef __attribute__((ext_vector_type(16))) _Float16 v16h;
typedef __attribute__((ext_vector_type(8)))  float    v8f;

#define NB   65536
#define SEQ  4
#define EMB  128

#define XP   130   // f32 row pitch (floats): bank advance 2
#define KP   132   // f16 row pitch (halfs):  264B = 66 words, bank advance 2

// prepacked weight offsets (in v16h units, 16 halfs each)
#define OFS_PROJ 0      // 4 matrices x 1024
#define OFS_WQ   4096
#define OFS_WK   5120
#define OFS_WV   6144
#define OFS_OUT  7168
#define OFS_GATE 8192   // 2048 (K=256 -> 8 kc)
#define PRE_TOTAL 10240 // v16h entries = 327680 bytes in d_ws

// per-wave LDS carve (bytes)
#define LO_X    0        // f32[4*16*XP]   33280
#define LO_Q    33280    // f32[16*XP]      8320
#define LO_K    41600    // f16[4*16*KP]   16896
#define LO_V    58496    // f16[4*16*KP]   16896
#define LO_O    75392    // f16[16*KP]      4224
#define LO_T    79616    // f32[64]
#define LO_CW   79872    // f32[64]
#define LO_AT   80128    // f32[256]
#define LO_RED  81152    // f32[64]
#define LO_MU   81408    // f32[16]
#define LO_RV   81472    // f32[16]
#define WAVE_BYTES 81536 // x2 waves = 163072 bytes/block (<= 160KB)

__device__ __forceinline__ v8f wmma16(v16h a, v16h b, v8f c) {
  return __builtin_amdgcn_wmma_f32_16x16x32_f16(false, a, false, b, (short)0, c,
                                                false, false);
}

// A-matrix 16x32 f16 fragment from an f32 matrix, row stride ld (floats).
// ISA 7.12.2: lanes 0-15 row=lane K=[0..7],[16..23]; lanes 16-31 K=[8..15],[24..31].
__device__ __forceinline__ v16h afrag(const float* __restrict__ p, int ld, int k0,
                                      int lane) {
  const int row = lane & 15;
  const int kb  = (lane & 16) ? 8 : 0;
  const float* q = p + row * ld + k0 + kb;
  v16h f;
#pragma unroll
  for (int j = 0; j < 8; ++j) f[j] = (_Float16)q[j];
#pragma unroll
  for (int j = 0; j < 8; ++j) f[j + 8] = (_Float16)q[16 + j];
  return f;
}

// Same but from an f16 source (no conversion needed).
__device__ __forceinline__ v16h afrag_h(const _Float16* __restrict__ p, int ld,
                                        int k0, int lane) {
  const int row = lane & 15;
  const int kb  = (lane & 16) ? 8 : 0;
  const _Float16* q = p + row * ld + k0 + kb;
  v16h f;
#pragma unroll
  for (int j = 0; j < 8; ++j) f[j] = q[j];
#pragma unroll
  for (int j = 0; j < 8; ++j) f[j + 8] = q[16 + j];
  return f;
}

// ---------------------------------------------------------------------------
// Prepack: convert f32 weights -> f16 B-fragment layout in d_ws.
// B-matrix 32x16 layout: lane holds column (lane&15); lanes<16 K=[0..7],[16..23],
// lanes>=16 K=[8..15],[24..31] (relative to kc*32), matching the A fragment.
// Stored at matbase + (nt*KC + kc)*32 + lane.
// ---------------------------------------------------------------------------
__global__ __launch_bounds__(256) void prepack_weights(
    const float* __restrict__ proj_w, const float* __restrict__ in_proj_w,
    const float* __restrict__ out_w, const float* __restrict__ gate_w,
    v16h* __restrict__ pre) {
  int id = blockIdx.x * blockDim.x + threadIdx.x;
  if (id >= PRE_TOTAL) return;

  const float* src;
  int rs, co, KC, r, matbase;
  if (id < 8192) {
    int m = id >> 10;
    r = id & 1023;
    KC = 4;
    matbase = m << 10;
    if (m < 4)        { src = proj_w + m * 128 * 128; rs = 128; co = 0;   }
    else if (m == 4)  { src = in_proj_w;              rs = 384; co = 0;   }
    else if (m == 5)  { src = in_proj_w;              rs = 384; co = 128; }
    else if (m == 6)  { src = in_proj_w;              rs = 384; co = 256; }
    else              { src = out_w;                  rs = 128; co = 0;   }
  } else {
    r = id - 8192; KC = 8; matbase = OFS_GATE; src = gate_w; rs = 128; co = 0;
  }
  int lane = r & 31;
  int tile = r >> 5;
  int kc = (KC == 4) ? (tile & 3) : (tile & 7);
  int col = co + ((KC == 4) ? (tile >> 2) : (tile >> 3)) * 16 + (lane & 15);
  int k0 = kc * 32 + ((lane & 16) ? 8 : 0);
  v16h f;
#pragma unroll
  for (int j = 0; j < 16; ++j) f[j] = (_Float16)src[(size_t)(k0 + ((j < 8) ? j : j + 8)) * rs + col];
  pre[matbase + r] = f;
}

// ---------------------------------------------------------------------------
__global__ __launch_bounds__(64) void mhf_fused(
    const float* __restrict__ h0, const float* __restrict__ h1,
    const float* __restrict__ h2, const float* __restrict__ h3,
    const float* __restrict__ proj_b, const float* __restrict__ ctx_w,
    const float* __restrict__ ctx_b, const float* __restrict__ in_proj_b,
    const float* __restrict__ out_b, const float* __restrict__ gate_b,
    const float* __restrict__ pool_w, const float* __restrict__ pool_b,
    const float* __restrict__ ln_g, const float* __restrict__ ln_b,
    const v16h* __restrict__ pre, float* __restrict__ out_pooled,
    float* __restrict__ out_attnw, float* __restrict__ out_g) {
  extern __shared__ char smraw[];
  const int wave = threadIdx.x >> 5;
  const int lane = threadIdx.x & 31;
  char* wb = smraw + (size_t)wave * WAVE_BYTES;
  float*    xls = (float*)(wb + LO_X);
  float*    qls = (float*)(wb + LO_Q);
  _Float16* kls = (_Float16*)(wb + LO_K);
  _Float16* vls = (_Float16*)(wb + LO_V);
  _Float16* ols = (_Float16*)(wb + LO_O);
  float*    tls = (float*)(wb + LO_T);
  float*    cwl = (float*)(wb + LO_CW);
  float*    atl = (float*)(wb + LO_AT);
  float*    red = (float*)(wb + LO_RED);
  float*    mul = (float*)(wb + LO_MU);
  float*    rvl = (float*)(wb + LO_RV);

  const int  ll = lane & 15;
  const int  hi = lane >> 4;
  const long base = (long)(blockIdx.x * 2 + wave) * 16;

  // ---- Phase A: x[s] = h[s](16x128) @ proj_w[s] + proj_b[s] -> LDS (f32) ----
  // Unrolled so each h pointer is a direct kernarg (global_load, not flat).
#pragma unroll
  for (int s = 0; s < 4; ++s) {
    const float* hp = (s == 0) ? h0 : (s == 1) ? h1 : (s == 2) ? h2 : h3;
    v16h a[4];
#pragma unroll
    for (int kc = 0; kc < 4; ++kc) a[kc] = afrag(hp + base * 128, 128, kc * 32, lane);
#pragma unroll
    for (int nt = 0; nt < 8; ++nt) {
      v8f acc = {};
#pragma unroll
      for (int kc = 0; kc < 4; ++kc)
        acc = wmma16(a[kc], pre[OFS_PROJ + s * 1024 + (nt * 4 + kc) * 32 + lane], acc);
      int col = nt * 16 + ll;
      float bias = proj_b[s * 128 + col];
#pragma unroll
      for (int r = 0; r < 8; ++r)
        xls[(s * 16 + r + hi * 8) * XP + col] = acc[r] + bias;
    }
  }
  __syncthreads();

  // ---- Phase B: context logits + softmax over s ----
  {
    float cb = ctx_b[0];
#pragma unroll
    for (int pp = 0; pp < 2; ++pp) {
      int p = lane + pp * 32;
      int row = p & 15, s = p >> 4;
      float t = cb;
      for (int c = 0; c < 128; ++c) t += xls[(s * 16 + row) * XP + c] * ctx_w[c];
      tls[s * 16 + row] = t;
    }
  }
  __syncthreads();
  if (lane < 16) {
    int row = lane;
    float m = tls[row];
#pragma unroll
    for (int s = 1; s < 4; ++s) m = fmaxf(m, tls[s * 16 + row]);
    float e[4], sum = 0.f;
#pragma unroll
    for (int s = 0; s < 4; ++s) { e[s] = expf(tls[s * 16 + row] - m); sum += e[s]; }
    float inv = 1.f / sum;
#pragma unroll
    for (int s = 0; s < 4; ++s) cwl[s * 16 + row] = e[s] * inv;
  }
  __syncthreads();

  // ---- Phase C: context = sum_s cw*x ; q = context @ wq + bq -> qls ----
  {
    int row = ll;
    float cwr[4];
#pragma unroll
    for (int s = 0; s < 4; ++s) cwr[s] = cwl[s * 16 + row];
    v16h ca[4];
    int kb = hi ? 8 : 0;
#pragma unroll
    for (int kc = 0; kc < 4; ++kc) {
#pragma unroll
      for (int j = 0; j < 16; ++j) {
        int k = kc * 32 + kb + ((j < 8) ? j : j + 8);
        float v = cwr[0] * xls[row * XP + k] + cwr[1] * xls[(16 + row) * XP + k] +
                  cwr[2] * xls[(32 + row) * XP + k] + cwr[3] * xls[(48 + row) * XP + k];
        ca[kc][j] = (_Float16)v;
      }
    }
#pragma unroll
    for (int nt = 0; nt < 8; ++nt) {
      v8f acc = {};
#pragma unroll
      for (int kc = 0; kc < 4; ++kc)
        acc = wmma16(ca[kc], pre[OFS_WQ + (nt * 4 + kc) * 32 + lane], acc);
      int col = nt * 16 + ll;
      float bq = in_proj_b[col];
#pragma unroll
      for (int r = 0; r < 8; ++r) qls[(r + hi * 8) * XP + col] = acc[r] + bq;
    }
  }
  __syncthreads();

  // ---- Phase D: k[s], v[s] = x[s] @ wk/wv + bias -> LDS (f16) ----
  for (int s = 0; s < 4; ++s) {
    v16h a[4];
#pragma unroll
    for (int kc = 0; kc < 4; ++kc) a[kc] = afrag(xls + s * 16 * XP, XP, kc * 32, lane);
#pragma unroll
    for (int nt = 0; nt < 8; ++nt) {
      v8f ak = {}, av = {};
#pragma unroll
      for (int kc = 0; kc < 4; ++kc) {
        ak = wmma16(a[kc], pre[OFS_WK + (nt * 4 + kc) * 32 + lane], ak);
        av = wmma16(a[kc], pre[OFS_WV + (nt * 4 + kc) * 32 + lane], av);
      }
      int col = nt * 16 + ll;
      float bk = in_proj_b[128 + col], bv = in_proj_b[256 + col];
#pragma unroll
      for (int r = 0; r < 8; ++r) {
        int row = r + hi * 8;
        kls[(s * 16 + row) * KP + col] = (_Float16)(ak[r] + bk);
        vls[(s * 16 + row) * KP + col] = (_Float16)(av[r] + bv);
      }
    }
  }
  __syncthreads();

  // ---- Phase E: scores -> softmax over s -> attn; o = attn @ v ----
#pragma unroll
  for (int pp = 0; pp < 2; ++pp) {
    int p = lane + pp * 32;
    int row = p & 15, h = p >> 4;
    float sc[4];
#pragma unroll
    for (int s = 0; s < 4; ++s) {
      float d = 0.f;
      for (int dd = 0; dd < 32; ++dd)
        d += qls[row * XP + h * 32 + dd] *
             (float)kls[(s * 16 + row) * KP + h * 32 + dd];
      sc[s] = d * 0.17677669529663687f;  // 1/sqrt(32)
    }
    float m = fmaxf(fmaxf(sc[0], sc[1]), fmaxf(sc[2], sc[3]));
    float e[4], sum = 0.f;
#pragma unroll
    for (int s = 0; s < 4; ++s) { e[s] = expf(sc[s] - m); sum += e[s]; }
    float inv = 1.f / sum;
#pragma unroll
    for (int s = 0; s < 4; ++s) atl[(row * 4 + h) * 4 + s] = e[s] * inv;
  }
  __syncthreads();
  if (lane < 16) {
    int row = lane;
#pragma unroll
    for (int s = 0; s < 4; ++s) {
      float a4 = 0.f;
#pragma unroll
      for (int h = 0; h < 4; ++h) a4 += atl[(row * 4 + h) * 4 + s];
      out_attnw[(base + row) * 4 + s] = a4 * 0.25f;
    }
  }
  for (int e = 0; e < 64; ++e) {
    int idx = lane + e * 32;
    int row = idx >> 7, col = idx & 127, h = col >> 5;
    float o = 0.f;
#pragma unroll
    for (int s = 0; s < 4; ++s)
      o += atl[(row * 4 + h) * 4 + s] * (float)vls[(s * 16 + row) * KP + col];
    ols[row * KP + col] = (_Float16)o;
  }
  __syncthreads();

  // ---- Phase F: attn_out = o @ out_w + out_b -> reuse qls as "ao" ----
  {
    v16h a[4];
#pragma unroll
    for (int kc = 0; kc < 4; ++kc) a[kc] = afrag_h(ols, KP, kc * 32, lane);
#pragma unroll
    for (int nt = 0; nt < 8; ++nt) {
      v8f acc = {};
#pragma unroll
      for (int kc = 0; kc < 4; ++kc)
        acc = wmma16(a[kc], pre[OFS_OUT + (nt * 4 + kc) * 32 + lane], acc);
      int col = nt * 16 + ll;
      float bo = out_b[col];
#pragma unroll
      for (int r = 0; r < 8; ++r) qls[(r + hi * 8) * XP + col] = acc[r] + bo;
    }
  }
  __syncthreads();

  // ---- Phase G: gate (16x256 @ 256x128), fused, LayerNorm (in-place in xls) ----
  for (int s = 0; s < 4; ++s) {
    v16h ga[8];
#pragma unroll
    for (int kc = 0; kc < 4; ++kc) ga[kc] = afrag(xls + s * 16 * XP, XP, kc * 32, lane);
#pragma unroll
    for (int kc = 0; kc < 4; ++kc) ga[4 + kc] = afrag(qls, XP, kc * 32, lane);
#pragma unroll
    for (int nt = 0; nt < 8; ++nt) {
      v8f acc = {};
#pragma unroll
      for (int kc = 0; kc < 8; ++kc)
        acc = wmma16(ga[kc], pre[OFS_GATE + (nt * 8 + kc) * 32 + lane], acc);
      int col = nt * 16 + ll;
      float gb = gate_b[col];
#pragma unroll
      for (int r = 0; r < 8; ++r) {
        int row = r + hi * 8;
        float gv = 1.f / (1.f + expf(-(acc[r] + gb)));
        float xv = xls[(s * 16 + row) * XP + col];
        float av = qls[row * XP + col];
        out_g[((base + row) * 4 + s) * 128 + col] = gv;
        xls[(s * 16 + row) * XP + col] = gv * av + (1.f - gv) * xv;
      }
    }
    __syncthreads();
    // LayerNorm stats: lane pair (row=ll, half=hi) each sums 64 cols
    {
      float sum = 0.f, sq = 0.f;
      for (int c = 0; c < 64; ++c) {
        float v = xls[(s * 16 + ll) * XP + hi * 64 + c];
        sum += v; sq += v * v;
      }
      red[lane] = sum;
      red[32 + lane] = sq;
    }
    __syncthreads();
    if (lane < 16) {
      float tot = red[lane] + red[lane + 16];
      float tsq = red[32 + lane] + red[48 + lane];
      float mu = tot * (1.f / 128.f);
      float var = tsq * (1.f / 128.f) - mu * mu;
      mul[lane] = mu;
      rvl[lane] = rsqrtf(var + 1e-5f);
    }
    __syncthreads();
    for (int e = 0; e < 64; ++e) {
      int idx = lane + e * 32;
      int row = idx >> 7, col = idx & 127;
      float v = xls[(s * 16 + row) * XP + col];
      xls[(s * 16 + row) * XP + col] = (v - mul[row]) * rvl[row] * ln_g[col] + ln_b[col];
    }
    __syncthreads();
  }

  // ---- Phase H: pool softmax over s, pooled output ----
  {
    float pb = pool_b[0];
#pragma unroll
    for (int pp = 0; pp < 2; ++pp) {
      int p = lane + pp * 32;
      int row = p & 15, s = p >> 4;
      float t = pb;
      for (int c = 0; c < 128; ++c) t += xls[(s * 16 + row) * XP + c] * pool_w[c];
      tls[s * 16 + row] = t;
    }
  }
  __syncthreads();
  if (lane < 16) {
    int row = lane;
    float m = tls[row];
#pragma unroll
    for (int s = 1; s < 4; ++s) m = fmaxf(m, tls[s * 16 + row]);
    float e[4], sum = 0.f;
#pragma unroll
    for (int s = 0; s < 4; ++s) { e[s] = expf(tls[s * 16 + row] - m); sum += e[s]; }
    float inv = 1.f / sum;
#pragma unroll
    for (int s = 0; s < 4; ++s) cwl[s * 16 + row] = e[s] * inv;
  }
  __syncthreads();
  for (int e = 0; e < 64; ++e) {
    int idx = lane + e * 32;
    int row = idx >> 7, col = idx & 127;
    float pv = 0.f;
#pragma unroll
    for (int s = 0; s < 4; ++s)
      pv += cwl[s * 16 + row] * xls[(s * 16 + row) * XP + col];
    out_pooled[(base + row) * 128 + col] = pv;
  }
}

// ---------------------------------------------------------------------------
extern "C" void kernel_launch(void* const* d_in, const int* in_sizes, int n_in,
                              void* d_out, int out_size, void* d_ws, size_t ws_size,
                              hipStream_t stream) {
  (void)in_sizes; (void)n_in; (void)out_size; (void)ws_size;
  const float* h0        = (const float*)d_in[0];
  const float* h1        = (const float*)d_in[1];
  const float* h2        = (const float*)d_in[2];
  const float* h3        = (const float*)d_in[3];
  const float* proj_w    = (const float*)d_in[4];
  const float* proj_b    = (const float*)d_in[5];
  const float* ctx_w     = (const float*)d_in[6];
  const float* ctx_b     = (const float*)d_in[7];
  const float* in_proj_w = (const float*)d_in[8];
  const float* in_proj_b = (const float*)d_in[9];
  const float* out_w     = (const float*)d_in[10];
  const float* out_b     = (const float*)d_in[11];
  const float* gate_w    = (const float*)d_in[12];
  const float* gate_b    = (const float*)d_in[13];
  const float* pool_w    = (const float*)d_in[14];
  const float* pool_b    = (const float*)d_in[15];
  const float* ln_g      = (const float*)d_in[16];
  const float* ln_b      = (const float*)d_in[17];

  v16h* pre = (v16h*)d_ws;
  prepack_weights<<<(PRE_TOTAL + 255) / 256, 256, 0, stream>>>(proj_w, in_proj_w,
                                                               out_w, gate_w, pre);

  float* out_pooled = (float*)d_out;
  float* out_attnw  = out_pooled + (size_t)NB * EMB;
  float* out_g      = out_attnw + (size_t)NB * SEQ;

  mhf_fused<<<NB / 32, 64, 2 * WAVE_BYTES, stream>>>(
      h0, h1, h2, h3, proj_b, ctx_w, ctx_b, in_proj_b, out_b, gate_b, pool_w,
      pool_b, ln_g, ln_b, pre, out_pooled, out_attnw, out_g);
}
